// AttentionBlock_28089086116213
// MI455X (gfx1250) — matrix operand order repaired and checked by reading, not compiled, not run on hardware
//
#include <hip/hip_runtime.h>
#include <hip/hip_bf16.h>
#include <math.h>

typedef __bf16 bf16_t;
typedef __attribute__((ext_vector_type(8)))  __bf16 v8bf;
typedef __attribute__((ext_vector_type(16))) __bf16 v16bf;
typedef __attribute__((ext_vector_type(8)))  float  v8f;

#define B_  16
#define C_  256
#define N_  1024
#define NH_ 4
#define D_  64

// ---------------- WMMA fragment helpers (wave32, 16x16x32 bf16) ----------------
// A fragment: row-major [16 x K] tile. Lane l: row = l&15; holds K {k0..k0+15}, k0 = 16*(l>>4).
__device__ __forceinline__ v16bf load_frag_a(const bf16_t* p, int ld) {
  int lane = threadIdx.x & 31;
  const bf16_t* q = p + (lane & 15) * ld + ((lane >> 4) << 4);
  v8bf lo = *(const v8bf*)q;
  v8bf hi = *(const v8bf*)(q + 8);
  v16bf out;
#pragma unroll
  for (int i = 0; i < 8; ++i) { out[i] = lo[i]; out[i + 8] = hi[i]; }
  return out;
}

// B fragment from B^T stored row-major [16 cols x K]. Lane l: col = l&15; K {k0..k0+15}, k0 = 16*(l>>4).
__device__ __forceinline__ v16bf load_frag_b(const bf16_t* p, int ld) {
  int lane = threadIdx.x & 31;
  return *(const v16bf*)(p + (lane & 15) * ld + ((lane >> 4) << 4));
}

__device__ __forceinline__ v8f wmma_bf16(v16bf a, v16bf b, v8f c) {
  return __builtin_amdgcn_wmma_f32_16x16x32_bf16(false, a, false, b, (short)0, c, false, false);
}

// 32(M) x 64(N) tile per wave, K-loop in steps of 32.
template <int K>
__device__ __forceinline__ void wave_gemm_32x64(const bf16_t* A, int lda,
                                                const bf16_t* Bt, int ldb,
                                                v8f acc[2][4]) {
#pragma unroll 2
  for (int k = 0; k < K; k += 32) {
    v16bf a0 = load_frag_a(A + k, lda);
    v16bf a1 = load_frag_a(A + 16 * lda + k, lda);
#pragma unroll
    for (int t = 0; t < 4; ++t) {
      v16bf b = load_frag_b(Bt + (t * 16) * ldb + k, ldb);
      acc[0][t] = wmma_bf16(a0, b, acc[0][t]);
      acc[1][t] = wmma_bf16(a1, b, acc[1][t]);
    }
  }
}

// ---------------- Kernel 1: GroupNorm, write h^T [b, n, c] in bf16 ----------------
__global__ __launch_bounds__(256) void groupnorm_kernel(const float* __restrict__ x,
                                                        const float* __restrict__ gamma,
                                                        const float* __restrict__ beta,
                                                        bf16_t* __restrict__ h_t) {
  int bg = blockIdx.x;            // 16*32 blocks
  int b = bg >> 5, g = bg & 31;   // 32 groups of 8 channels
  const float* xp = x + ((size_t)b * C_ + g * 8) * N_;  // group = 8192 contiguous floats
  int tid = threadIdx.x;
  float s1 = 0.f, s2 = 0.f;
#pragma unroll
  for (int j = 0; j < 32; ++j) {
    float v = xp[tid + 256 * j];
    s1 += v; s2 += v * v;
  }
#pragma unroll
  for (int off = 16; off > 0; off >>= 1) {
    s1 += __shfl_xor(s1, off, 32);
    s2 += __shfl_xor(s2, off, 32);
  }
  __shared__ float red[2][8];
  if ((tid & 31) == 0) { red[0][tid >> 5] = s1; red[1][tid >> 5] = s2; }
  __syncthreads();
  float t1 = 0.f, t2 = 0.f;
#pragma unroll
  for (int w = 0; w < 8; ++w) { t1 += red[0][w]; t2 += red[1][w]; }
  float mean = t1 * (1.0f / 8192.0f);
  float var  = t2 * (1.0f / 8192.0f) - mean * mean;
  float rstd = rsqrtf(var + 1e-5f);
  float gm[8], bt[8];
#pragma unroll
  for (int cc = 0; cc < 8; ++cc) {
    gm[cc] = gamma[g * 8 + cc] * rstd;
    bt[cc] = beta[g * 8 + cc] - mean * gm[cc];
  }
  bf16_t* op = h_t + (size_t)b * N_ * C_ + g * 8;
#pragma unroll
  for (int j = 0; j < 4; ++j) {
    int n = tid + 256 * j;
    v8bf o;
#pragma unroll
    for (int cc = 0; cc < 8; ++cc)
      o[cc] = (bf16_t)(xp[cc * N_ + n] * gm[cc] + bt[cc]);
    *(v8bf*)(op + (size_t)n * C_) = o;
  }
}

// ---------------- Kernel 2: weights fp32 -> bf16 ----------------
__global__ void cvt_weights_kernel(const float* __restrict__ qkv_w, const float* __restrict__ proj_w,
                                   bf16_t* __restrict__ w_qkv, bf16_t* __restrict__ w_proj) {
  int idx = blockIdx.x * blockDim.x + threadIdx.x;
  if (idx < 3 * C_ * C_) w_qkv[idx] = (bf16_t)qkv_w[idx];
  if (idx < C_ * C_)     w_proj[idx] = (bf16_t)proj_w[idx];
}

// ---------------- Kernel 3: QKV GEMM [N,C]x[C,3C], scatter to q_t/k_t [b,h,n,d], v [b,h,d,n] ----------------
__global__ __launch_bounds__(128) void qkv_gemm_kernel(const bf16_t* __restrict__ h_t,
                                                       const bf16_t* __restrict__ w_qkv,
                                                       const float* __restrict__ qkv_b,
                                                       bf16_t* __restrict__ q_t,
                                                       bf16_t* __restrict__ k_t,
                                                       bf16_t* __restrict__ v_m) {
  int wid = blockIdx.x * (blockDim.x >> 5) + (threadIdx.x >> 5);
  int b = wid / (32 * 12);
  int rem = wid % (32 * 12);
  int m0 = (rem / 12) * 32;
  int o0 = (rem % 12) * 64;
  const bf16_t* A  = h_t + (size_t)b * N_ * C_ + (size_t)m0 * C_;
  const bf16_t* Bt = w_qkv + (size_t)o0 * C_;
  v8f acc[2][4] = {};
  wave_gemm_32x64<C_>(A, C_, Bt, C_, acc);

  int lane = threadIdx.x & 31;
  int col_l = lane & 15, hi = lane >> 4;
#pragma unroll
  for (int half = 0; half < 2; ++half)
#pragma unroll
    for (int t = 0; t < 4; ++t) {
      int o = o0 + t * 16 + col_l;
      float bias = qkv_b[o];
      int head = (o >> 6) & 3;
      int dd = o & 63;
      size_t hb = (size_t)b * NH_ + head;
#pragma unroll
      for (int i = 0; i < 8; ++i) {
        int n = m0 + half * 16 + hi * 8 + i;
        float val = acc[half][t][i] + bias;
        if (o < 256)      q_t[(hb * N_ + n) * D_ + dd] = (bf16_t)(val * 0.125f);  // scale folded
        else if (o < 512) k_t[(hb * N_ + n) * D_ + dd] = (bf16_t)val;
        else              v_m[(hb * D_ + dd) * N_ + n] = (bf16_t)val;             // transposed
      }
    }
}

// ---------------- Kernel 4: flash attention, one wave = 16 queries ----------------
// S^T[m,n] tiles: A = k_t rows(m), B^T = q_t rows(n). Softmax axis m is per-lane -> shfl_xor(16) only.
// AV: O^T[dd,n] with A = v rows(dd), B^T = P (built from exp'd S^T via 2 shuffles). Output -> [b,n,c].
__global__ __launch_bounds__(256) void attn_kernel(const bf16_t* __restrict__ q_t,
                                                   const bf16_t* __restrict__ k_t,
                                                   const bf16_t* __restrict__ v_m,
                                                   bf16_t* __restrict__ at) {
  int wid = blockIdx.x * (blockDim.x >> 5) + (threadIdx.x >> 5);
  int b = wid >> 8;
  int head = (wid >> 6) & 3;
  int n0 = (wid & 63) << 4;
  size_t hb = (size_t)b * NH_ + head;
  const bf16_t* qb = q_t + (hb * N_ + n0) * D_;
  const bf16_t* kb = k_t + hb * N_ * D_;
  const bf16_t* vb = v_m + hb * D_ * N_;

  v16bf q0 = load_frag_b(qb, D_);        // K = 0..31 of d
  v16bf q1 = load_frag_b(qb + 32, D_);   // K = 32..63
  v8f acc[4] = {};
  float m_run = -INFINITY, l_run = 0.f;
  int lane = threadIdx.x & 31;
  bool lolane = lane < 16;

  for (int m0 = 0; m0 < N_; m0 += 32) {
    v8f s0 = {}, s1 = {};
    s0 = wmma_bf16(load_frag_a(kb + (size_t)m0 * D_, D_),        q0, s0);
    s0 = wmma_bf16(load_frag_a(kb + (size_t)m0 * D_ + 32, D_),   q1, s0);
    s1 = wmma_bf16(load_frag_a(kb + (size_t)(m0 + 16) * D_, D_),      q0, s1);
    s1 = wmma_bf16(load_frag_a(kb + (size_t)(m0 + 16) * D_ + 32, D_), q1, s1);

    float cmax = -INFINITY;
#pragma unroll
    for (int i = 0; i < 8; ++i) cmax = fmaxf(cmax, fmaxf(s0[i], s1[i]));
    cmax = fmaxf(cmax, __shfl_xor(cmax, 16, 32));
    float m_new = fmaxf(m_run, cmax);
    float alpha = __expf(m_run - m_new);
    float p0[8], p1[8], lsum = 0.f;
#pragma unroll
    for (int i = 0; i < 8; ++i) {
      p0[i] = __expf(s0[i] - m_new);
      p1[i] = __expf(s1[i] - m_new);
      lsum += p0[i] + p1[i];
    }
    lsum += __shfl_xor(lsum, 16, 32);
    l_run = l_run * alpha + lsum;
    m_run = m_new;

    // Build P B-fragment (lane col n = lane&15; K = m-m0): lo lanes need m 0..15, hi lanes 16..31.
    v16bf pb;
#pragma unroll
    for (int i = 0; i < 8; ++i) {
      float o0v = __shfl_xor(p0[i], 16, 32);
      float o1v = __shfl_xor(p1[i], 16, 32);
      pb[i]     = (bf16_t)(lolane ? p0[i] : o1v);
      pb[i + 8] = (bf16_t)(lolane ? o0v   : p1[i]);
    }
#pragma unroll
    for (int t = 0; t < 4; ++t) acc[t] *= alpha;
#pragma unroll
    for (int t = 0; t < 4; ++t)
      acc[t] = wmma_bf16(load_frag_a(vb + (size_t)(t * 16) * N_ + m0, N_), pb, acc[t]);
  }

  float inv = 1.0f / l_run;
  int col_l = lane & 15, hi = lane >> 4;
  int n = n0 + col_l;
  bf16_t* outp = at + ((size_t)b * N_ + n) * C_ + head * D_ + hi * 8;
#pragma unroll
  for (int t = 0; t < 4; ++t) {
    v8bf o;
#pragma unroll
    for (int i = 0; i < 8; ++i) o[i] = (bf16_t)(acc[t][i] * inv);
    *(v8bf*)(outp + t * 16) = o;
  }
}

// ---------------- Kernel 5: proj GEMM + bias + residual -> fp32 out [b,o,n] ----------------
__global__ __launch_bounds__(128) void proj_gemm_kernel(const bf16_t* __restrict__ at,
                                                        const bf16_t* __restrict__ w_proj,
                                                        const float* __restrict__ proj_b,
                                                        const float* __restrict__ x,
                                                        float* __restrict__ out) {
  int wid = blockIdx.x * (blockDim.x >> 5) + (threadIdx.x >> 5);
  int b = wid / (32 * 4);
  int rem = wid % (32 * 4);
  int m0 = (rem / 4) * 32;
  int o0 = (rem % 4) * 64;
  const bf16_t* A  = at + (size_t)b * N_ * C_ + (size_t)m0 * C_;
  const bf16_t* Bt = w_proj + (size_t)o0 * C_;
  v8f acc[2][4] = {};
  wave_gemm_32x64<C_>(A, C_, Bt, C_, acc);

  int lane = threadIdx.x & 31;
  int col_l = lane & 15, hi = lane >> 4;
#pragma unroll
  for (int half = 0; half < 2; ++half)
#pragma unroll
    for (int t = 0; t < 4; ++t) {
      int o = o0 + t * 16 + col_l;
      float bias = proj_b[o];
      size_t base = ((size_t)b * C_ + o) * N_;
#pragma unroll
      for (int i = 0; i < 8; ++i) {
        int n = m0 + half * 16 + hi * 8 + i;
        out[base + n] = x[base + n] + acc[half][t][i] + bias;
      }
    }
}

extern "C" void kernel_launch(void* const* d_in, const int* in_sizes, int n_in,
                              void* d_out, int out_size, void* d_ws, size_t ws_size,
                              hipStream_t stream) {
  const float* x      = (const float*)d_in[0];
  const float* gamma  = (const float*)d_in[1];
  const float* beta   = (const float*)d_in[2];
  const float* qkv_w  = (const float*)d_in[3];
  const float* qkv_b  = (const float*)d_in[4];
  const float* proj_w = (const float*)d_in[5];
  const float* proj_b = (const float*)d_in[6];
  float* out = (float*)d_out;

  const size_t TN = (size_t)B_ * N_ * C_;  // 4,194,304
  bf16_t* ws    = (bf16_t*)d_ws;
  bf16_t* h_t   = ws;                      // [B, N, C]
  bf16_t* w_qkv = h_t + TN;                // [3C, C]
  bf16_t* w_prj = w_qkv + 3 * C_ * C_;     // [C, C]
  bf16_t* q_t   = w_prj + C_ * C_;         // [B, nh, N, d]
  bf16_t* k_t   = q_t + TN;                // [B, nh, N, d]
  bf16_t* v_m   = k_t + TN;                // [B, nh, d, N]
  bf16_t* at    = v_m + TN;                // [B, N, C]

  groupnorm_kernel<<<B_ * 32, 256, 0, stream>>>(x, gamma, beta, h_t);
  cvt_weights_kernel<<<(3 * C_ * C_ + 255) / 256, 256, 0, stream>>>(qkv_w, proj_w, w_qkv, w_prj);
  qkv_gemm_kernel<<<B_ * 32 * 12 / 4, 128, 0, stream>>>(h_t, w_qkv, qkv_b, q_t, k_t, v_m);
  attn_kernel<<<B_ * NH_ * 64 / 8, 256, 0, stream>>>(q_t, k_t, v_m, at);
  proj_gemm_kernel<<<B_ * 32 * 4 / 4, 128, 0, stream>>>(at, w_prj, proj_b, x, out);
}
